// PrototypeLearnerV4_62689342653058
// MI455X (gfx1250) — compile-verified, matrix operand order checked by reading, not run on hardware
//
#include <hip/hip_runtime.h>

typedef __bf16 bf16;
typedef bf16  v16bf  __attribute__((ext_vector_type(16)));
typedef bf16  bf16x8 __attribute__((ext_vector_type(8)));
typedef float v8f    __attribute__((ext_vector_type(8)));
typedef int   v4i    __attribute__((ext_vector_type(4)));

#if defined(__gfx1250__) && __has_builtin(__builtin_amdgcn_global_load_async_to_lds_b128)
#define USE_ASYNC_LDS 1
typedef __attribute__((address_space(1))) v4i* g_v4i_ptr;   // global int4*
typedef __attribute__((address_space(3))) v4i* l_v4i_ptr;   // LDS int4*
#else
#define USE_ASYNC_LDS 0
#endif

__device__ __forceinline__ bf16 f2bf(float f) {
    unsigned u = __builtin_bit_cast(unsigned, f);
    u += 0x7FFFu + ((u >> 16) & 1u);           // round-to-nearest-even
    unsigned short s = (unsigned short)(u >> 16);
    return __builtin_bit_cast(bf16, s);
}

// 16-byte global -> LDS copy: async DMA path on gfx1250 when available.
__device__ __forceinline__ void cp16_g2l(const bf16* g, bf16* l) {
#if USE_ASYNC_LDS
    __builtin_amdgcn_global_load_async_to_lds_b128(
        (g_v4i_ptr)(bf16*)g, (l_v4i_ptr)l, 0, 0);
#else
    *(bf16x8*)l = *(const bf16x8*)g;
#endif
}

// Must precede the barrier signal so that barrier completion implies the
// async LDS writes are visible (ASYNCcnt is not covered by __syncthreads).
__device__ __forceinline__ void wait_async_lds() {
#if USE_ASYNC_LDS
#if __has_builtin(__builtin_amdgcn_s_wait_asynccnt)
    __builtin_amdgcn_s_wait_asynccnt(0);
#else
    asm volatile("s_wait_asynccnt 0x0" ::: "memory");
#endif
#endif
}

// ---------------------------------------------------------------------------
// Convert fp32 [N x K] -> bf16 [N x Kpad], zero-padding the K tail.
// ---------------------------------------------------------------------------
__global__ void cvt_bf16_kernel(const float* __restrict__ W, bf16* __restrict__ Wb,
                                int N, int K, int Kpad) {
    int i = blockIdx.x * blockDim.x + threadIdx.x;
    int total = N * Kpad;
    if (i >= total) return;
    int n = i / Kpad, k = i - n * Kpad;
    float v = (k < K) ? W[(size_t)n * K + k] : 0.0f;
    Wb[i] = f2bf(v);
}

__global__ void zero_bf16_kernel(bf16* __restrict__ p, int n) {
    int i = blockIdx.x * blockDim.x + threadIdx.x;
    if (i < n) p[i] = __builtin_bit_cast(bf16, (unsigned short)0);
}

// ---------------------------------------------------------------------------
// GEMM: D = act( A @ Wb^T + bias ), A bf16 [M x K] (K % 32 == 0, pre-padded),
// Wb bf16 [N x K].  Epilogue optionally writes fp32 out and/or the next
// layer's bf16 input (with optional fp32 residual fused in).
//
// Block: 256 threads = 8 wave32.  Block tile: 256 rows x 16 cols.
// Each wave owns 32 rows = two 16x16 accumulators sharing one B fragment
// -> 2x v_wmma_f32_16x16x32_bf16 per K-step.  LDS tiles double-buffered
// (filled by async global->LDS DMA when available), one barrier per K-step,
// +2-step-ahead global_prefetch.
// ---------------------------------------------------------------------------
__global__ __launch_bounds__(256)
void gemm_bf16_wmma_kernel(const bf16* __restrict__ A, const bf16* __restrict__ Wb,
                           const float* __restrict__ bias,
                           float* __restrict__ out32, bf16* __restrict__ out16,
                           const float* __restrict__ res,
                           int N, int K, int strideB, int relu) {
    __shared__ __align__(16) bf16 sW[2][16 * 32];       // weight tile (double buffered)
    __shared__ __align__(16) bf16 sA[2][8][32 * 32];    // per-wave 32-row A tiles

    const int wave    = threadIdx.x >> 5;
    const int lane    = threadIdx.x & 31;
    const int rowBase = blockIdx.x * 256 + wave * 32;
    const int n0      = blockIdx.y * 16;
    const int nsteps  = K >> 5;

    const int wr = lane >> 1;            // W-loader: tile row 0..15
    const int wh = (lane & 1) * 16;      // W-loader: k-half 0/16

    auto loadW = [&](int buf, int k0) {
        if (wave == 0) {
            const bf16* src = Wb + (size_t)(n0 + wr) * K + k0 + wh;
            bf16* dst = &sW[buf][wr * 32 + wh];
            cp16_g2l(src,     dst);
            cp16_g2l(src + 8, dst + 8);
        }
    };
    auto loadA = [&](int buf, int k0) {  // lane moves one full 32-k row (64B)
        const bf16* src = A + (size_t)(rowBase + lane) * K + k0;
        bf16* dst = &sA[buf][wave][lane * 32];
        cp16_g2l(src,      dst);
        cp16_g2l(src + 8,  dst + 8);
        cp16_g2l(src + 16, dst + 16);
        cp16_g2l(src + 24, dst + 24);
    };

    v8f acc0 = {}, acc1 = {};
    loadW(0, 0);
    loadA(0, 0);

    const int mrow = lane & 15;          // fragment row (m or n)
    const int kg   = lane >> 4;          // k-group per ISA 16-bit layout

    for (int s = 0; s < nsteps; ++s) {
        const int cur = s & 1;
        wait_async_lds();                             // ASYNCcnt drained before signal
        __syncthreads();                              // buf 'cur' ready
        if (s + 1 < nsteps) {
            loadW(cur ^ 1, (s + 1) * 32);             // pipeline next tile
            loadA(cur ^ 1, (s + 1) * 32);
            if (s + 2 < nsteps) {                     // gfx1250 global_prefetch_b8
                __builtin_prefetch(Wb + (size_t)(n0 + wr) * K + (s + 2) * 32, 0, 1);
                __builtin_prefetch(A + (size_t)(rowBase + lane) * K + (s + 2) * 32, 0, 1);
            }
        }

        // Fragment gather per ISA 16-bit A/B layout: lane (row = lane%16,
        // kg = lane/16) holds k in [8*kg, 8*kg+8) and [16+8*kg, 24+8*kg).
        const bf16* br  = &sW[cur][mrow * 32];
        const bf16* ar0 = &sA[cur][wave][mrow * 32];
        const bf16* ar1 = &sA[cur][wave][(16 + mrow) * 32];
        bf16x8 b_lo  = *(const bf16x8*)(br  + kg * 8);
        bf16x8 b_hi  = *(const bf16x8*)(br  + 16 + kg * 8);
        bf16x8 a0_lo = *(const bf16x8*)(ar0 + kg * 8);
        bf16x8 a0_hi = *(const bf16x8*)(ar0 + 16 + kg * 8);
        bf16x8 a1_lo = *(const bf16x8*)(ar1 + kg * 8);
        bf16x8 a1_hi = *(const bf16x8*)(ar1 + 16 + kg * 8);
        v16bf af0, af1, bfr;
        #pragma unroll
        for (int i = 0; i < 8; ++i) {
            bfr[i] = b_lo[i];  bfr[8 + i] = b_hi[i];
            af0[i] = a0_lo[i]; af0[8 + i] = a0_hi[i];
            af1[i] = a1_lo[i]; af1[8 + i] = a1_hi[i];
        }
        acc0 = __builtin_amdgcn_wmma_f32_16x16x32_bf16(false, af0, false, bfr,
                                                       (short)0, acc0, false, false);
        acc1 = __builtin_amdgcn_wmma_f32_16x16x32_bf16(false, af1, false, bfr,
                                                       (short)0, acc1, false, false);
    }

    // Epilogue per C/D layout: VGPR r -> row (r + 8*(lane/16)), col = lane%16
    const int n  = n0 + mrow;
    const int mo = kg * 8;
    const float bv = bias ? bias[n] : 0.0f;
    #pragma unroll
    for (int t = 0; t < 2; ++t) {
        const v8f& acc = t ? acc1 : acc0;
        #pragma unroll
        for (int r = 0; r < 8; ++r) {
            const int row = rowBase + t * 16 + mo + r;
            float v = acc[r] + bv;
            if (relu) v = fmaxf(v, 0.0f);
            if (out32) out32[(size_t)row * N + n] = v;
            if (out16) {
                float vb = v;
                if (res) vb += res[(size_t)row * N + n];   // fused residual
                out16[(size_t)row * strideB + n] = f2bf(vb);
            }
        }
    }
}

// ---------------------------------------------------------------------------
// logits[b,p] = -sum_d |x[b,d] + y[b,d] + u[b,d] - protos[p,d]|
// ---------------------------------------------------------------------------
__global__ __launch_bounds__(256)
void l1_logits_kernel(const float* __restrict__ x, const float* __restrict__ y,
                      const float* __restrict__ u, const float* __restrict__ protos,
                      float* __restrict__ out, int D, int P) {
    __shared__ __align__(16) float feat[16 * 768];
    const int t = threadIdx.x;
    const size_t base = (size_t)blockIdx.x * 16 * D;
    for (int i = t; i < 16 * D; i += 256)
        feat[i] = x[base + i] + y[base + i] + u[base + i];
    __syncthreads();

    const int r  = t & 15;
    const int ps = t >> 4;
    const float* frow = &feat[r * D];
    for (int pb = 0; pb < P; pb += 16) {
        int p = pb + ps;
        if (p < P) {
            const float* pr = protos + (size_t)p * D;
            float acc = 0.0f;
            #pragma unroll 4
            for (int d = 0; d < D; d += 4) {
                float4 f = *(const float4*)(frow + d);
                float4 q = *(const float4*)(pr + d);
                acc += fabsf(f.x - q.x) + fabsf(f.y - q.y) +
                       fabsf(f.z - q.z) + fabsf(f.w - q.w);
            }
            out[(size_t)(blockIdx.x * 16 + r) * P + p] = -acc;
        }
    }
}

// ---------------------------------------------------------------------------
extern "C" void kernel_launch(void* const* d_in, const int* in_sizes, int n_in,
                              void* d_out, int out_size, void* d_ws, size_t ws_size,
                              hipStream_t stream) {
    const int B = 2048, D = 768, P = 200;

    const float* x      = (const float*)d_in[0];
    const float* aW     = (const float*)d_in[1];
    const float* ab     = (const float*)d_in[2];
    const float* protos = (const float*)d_in[3];
    const float* Wd[4]  = {(const float*)d_in[4],  (const float*)d_in[6],
                           (const float*)d_in[8],  (const float*)d_in[10]};
    const float* bd[4]  = {(const float*)d_in[5],  (const float*)d_in[7],
                           (const float*)d_in[9],  (const float*)d_in[11]};
    const float* Wu[4]  = {(const float*)d_in[12], (const float*)d_in[14],
                           (const float*)d_in[16], (const float*)d_in[18]};
    const float* bu[4]  = {(const float*)d_in[13], (const float*)d_in[15],
                           (const float*)d_in[17], (const float*)d_in[19]};

    char* w = (char*)d_ws;
    auto carve = [&](size_t bytes) { char* p = w; w += (bytes + 255) & ~(size_t)255; return p; };

    // bf16 weights (N, K, Kpad) — K padded to multiple of 32
    struct WSpec { const float* src; int N, K, Kpad; bf16* dst; };
    WSpec wsw[9] = {
        {aW,    768, 768, 768, nullptr},
        {Wd[0], 384, 768, 768, nullptr}, {Wd[1], 192, 384, 384, nullptr},
        {Wd[2],  96, 192, 192, nullptr}, {Wd[3],  48,  96,  96, nullptr},
        {Wu[0],  96,  48,  64, nullptr}, {Wu[1], 192,  96,  96, nullptr},
        {Wu[2], 384, 192, 192, nullptr}, {Wu[3], 768, 384, 384, nullptr},
    };
    for (int i = 0; i < 9; ++i)
        wsw[i].dst = (bf16*)carve((size_t)wsw[i].N * wsw[i].Kpad * sizeof(bf16));

    // bf16 activations (row stride = padded width)
    bf16* xb   = (bf16*)carve((size_t)B * 768 * 2);
    bf16* h1b  = (bf16*)carve((size_t)B * 384 * 2);
    bf16* h2b  = (bf16*)carve((size_t)B * 192 * 2);
    bf16* h3b  = (bf16*)carve((size_t)B *  96 * 2);
    bf16* h4b  = (bf16*)carve((size_t)B *  64 * 2);   // 48 real cols, padded to 64
    bf16* in2b = (bf16*)carve((size_t)B *  96 * 2);   // bf16(u1 + h3)
    bf16* in3b = (bf16*)carve((size_t)B * 192 * 2);   // bf16(u2 + h2)
    bf16* u3b  = (bf16*)carve((size_t)B * 384 * 2);

    // fp32 tensors still needed downstream
    float* y   = (float*)carve((size_t)B * 768 * 4);  // x@aW^T + ab
    float* h2f = (float*)carve((size_t)B * 192 * 4);  // residual source
    float* h3f = (float*)carve((size_t)B *  96 * 4);  // residual source
    float* u4f = (float*)carve((size_t)B * 768 * 4);  // adapter-net output

    // ---- conversions / init ----
    for (int i = 0; i < 9; ++i) {
        int total = wsw[i].N * wsw[i].Kpad;
        cvt_bf16_kernel<<<(total + 255) / 256, 256, 0, stream>>>(
            wsw[i].src, wsw[i].dst, wsw[i].N, wsw[i].K, wsw[i].Kpad);
    }
    cvt_bf16_kernel<<<(B * 768 + 255) / 256, 256, 0, stream>>>(x, xb, B, 768, 768);
    zero_bf16_kernel<<<(B * 64 + 255) / 256, 256, 0, stream>>>(h4b, B * 64);

    // ---- GEMM chain (all K pre-padded to multiples of 32) ----
    auto gemm = [&](const bf16* A, int widx, const float* bias,
                    float* o32, bf16* o16, int strideB, const float* res, int relu) {
        const int N = wsw[widx].N, K = wsw[widx].Kpad;
        dim3 g(B / 256, N / 16);
        gemm_bf16_wmma_kernel<<<g, 256, 0, stream>>>(
            A, wsw[widx].dst, bias, o32, o16, res, N, K, strideB, relu);
    };

    gemm(xb,   0, ab,    y,    nullptr, 0,   nullptr, 0); // y  = x@aW^T + ab
    gemm(xb,   1, bd[0], nullptr, h1b,  384, nullptr, 1); // h1
    gemm(h1b,  2, bd[1], h2f,  h2b,     192, nullptr, 1); // h2 (fp32 kept for residual)
    gemm(h2b,  3, bd[2], h3f,  h3b,      96, nullptr, 1); // h3 (fp32 kept for residual)
    gemm(h3b,  4, bd[3], nullptr, h4b,   64, nullptr, 1); // h4 (padded stride 64)
    gemm(h4b,  5, bu[0], nullptr, in2b,  96, h3f,     1); // in2 = bf16(u1 + h3)
    gemm(in2b, 6, bu[1], nullptr, in3b, 192, h2f,     1); // in3 = bf16(u2 + h2)
    gemm(in3b, 7, bu[2], nullptr, u3b,  384, nullptr, 1); // u3
    gemm(u3b,  8, bu[3], u4f,  nullptr, 0,   nullptr, 1); // u4 = adapter output

    // ---- features + L1 prototype distances ----
    l1_logits_kernel<<<B / 16, 256, 0, stream>>>(x, y, u4f, protos, (float*)d_out, D, P);

    (void)in_sizes; (void)n_in; (void)out_size; (void)ws_size;
}
